// Nordic4BModel_56873956934230
// MI455X (gfx1250) — matrix operand order repaired and checked by reading, not compiled, not run on hardware
//
#include <hip/hip_runtime.h>
#include <hip/hip_bf16.h>
#include <cstdint>

// ---------------- model constants ----------------
constexpr int Vv = 32000, Dd = 1024, Hh = 16, Gg = 4, HD = 64, Ff = 4096, Ll = 8;
constexpr int Ss = 2048, Bb = 2, Tt = Bb * Ss;
constexpr int WINDOW = 1024;
constexpr int GROUP = Hh / Gg;
constexpr float EPS = 1e-6f;

typedef __attribute__((ext_vector_type(16))) _Float16 v16h;
typedef __attribute__((ext_vector_type(8)))  float    v8f;
typedef __attribute__((ext_vector_type(4)))  float    v4f;

union Frag {
    v16h      v;
    _Float16  h[16];
    uint32_t  u[8];
    v4f       f4[2];
};

// ---------------- f32 -> f16 convert (no transpose; used for tok_emb) ----------------
__global__ __launch_bounds__(256) void nordic_conv_kernel(const float* __restrict__ in,
                                                          _Float16* __restrict__ out, size_t n) {
    size_t i = (size_t)blockIdx.x * 256 + threadIdx.x;
    if (i < n) out[i] = (_Float16)in[i];
}

// ---------------- f32 [K x N] -> f16 [N x K] transpose-convert, batched over layers ----------------
__global__ __launch_bounds__(256) void nordic_convT_kernel(const float* __restrict__ in,
                                                           _Float16* __restrict__ out,
                                                           int K, int N) {
    __shared__ _Float16 tile[32][33];
    const float* inz = in + (size_t)blockIdx.z * K * N;
    _Float16*    outz = out + (size_t)blockIdx.z * K * N;
    int n0 = blockIdx.x * 32, k0 = blockIdx.y * 32;
    int tx = threadIdx.x & 31, ty = threadIdx.x >> 5; // 32 x 8
#pragma unroll
    for (int i = 0; i < 32; i += 8)
        tile[ty + i][tx] = (_Float16)inz[(size_t)(k0 + ty + i) * N + n0 + tx];
    __syncthreads();
#pragma unroll
    for (int i = 0; i < 32; i += 8)
        outz[(size_t)(n0 + ty + i) * K + k0 + tx] = tile[tx][ty + i];
}

// ---------------- embedding gather ----------------
__global__ __launch_bounds__(256) void nordic_embed_kernel(const int* __restrict__ ids,
                                                           const float* __restrict__ emb,
                                                           float* __restrict__ x) {
    size_t i = (size_t)blockIdx.x * 256 + threadIdx.x;
    if (i >= (size_t)Tt * Dd) return;
    int t = (int)(i / Dd), d = (int)(i % Dd);
    x[i] = emb[(size_t)ids[t] * Dd + d] * 32.0f; // sqrt(1024)
}

// ---------------- rmsnorm (row of D) -> f16 ----------------
__global__ __launch_bounds__(256) void nordic_rmsnorm_kernel(const float* __restrict__ x,
                                                             const float* __restrict__ w,
                                                             _Float16* __restrict__ out) {
    int row = blockIdx.x;
    const float* xr = x + (size_t)row * Dd;
    float s = 0.f;
    for (int d = threadIdx.x; d < Dd; d += 256) { float v = xr[d]; s += v * v; }
    __shared__ float red[256];
    red[threadIdx.x] = s; __syncthreads();
    for (int off = 128; off > 0; off >>= 1) {
        if (threadIdx.x < off) red[threadIdx.x] += red[threadIdx.x + off];
        __syncthreads();
    }
    float rs = rsqrtf(red[0] / (float)Dd + EPS);
    for (int d = threadIdx.x; d < Dd; d += 256)
        out[(size_t)row * Dd + d] = (_Float16)(xr[d] * rs * (1.0f + w[d]));
}

// ---------------- x += rmsnorm(y, w) ----------------
__global__ __launch_bounds__(256) void nordic_addnorm_kernel(float* __restrict__ x,
                                                             const float* __restrict__ y,
                                                             const float* __restrict__ w) {
    int row = blockIdx.x;
    const float* yr = y + (size_t)row * Dd;
    float s = 0.f;
    for (int d = threadIdx.x; d < Dd; d += 256) { float v = yr[d]; s += v * v; }
    __shared__ float red[256];
    red[threadIdx.x] = s; __syncthreads();
    for (int off = 128; off > 0; off >>= 1) {
        if (threadIdx.x < off) red[threadIdx.x] += red[threadIdx.x + off];
        __syncthreads();
    }
    float rs = rsqrtf(red[0] / (float)Dd + EPS);
    for (int d = threadIdx.x; d < Dd; d += 256)
        x[(size_t)row * Dd + d] += yr[d] * rs * (1.0f + w[d]);
}

// ---- per-head rmsnorm + rope; in (t,nh,HD) f32; out f16:
//      transpose_out==0 -> (b,nh,s,HD);  transpose_out==1 -> (b,nh,HD,s) ----
__global__ __launch_bounds__(256) void nordic_qkrope_kernel(const float* __restrict__ in,
                                                            const float* __restrict__ w,   // null -> no norm
                                                            const float* __restrict__ cosT,
                                                            const float* __restrict__ sinT,
                                                            _Float16* __restrict__ out,
                                                            int nh, float scale, int do_rope,
                                                            int transpose_out) {
    int wid  = blockIdx.x * 8 + (threadIdx.x >> 5);
    int lane = threadIdx.x & 31;
    int rows = Tt * nh;
    if (wid >= rows) return;
    int t = wid / nh, hh = wid % nh;
    int b = t / Ss, s = t % Ss;
    const float* xr = in + (size_t)(t * nh + hh) * HD;
    float x0 = xr[lane], x1 = xr[lane + 32];
    float xn0 = x0, xn1 = x1;
    if (w) {
        float ss = x0 * x0 + x1 * x1;
        ss += __shfl_xor(ss, 1, 32);  ss += __shfl_xor(ss, 2, 32);
        ss += __shfl_xor(ss, 4, 32);  ss += __shfl_xor(ss, 8, 32);
        ss += __shfl_xor(ss, 16, 32);
        float rs = rsqrtf(ss / (float)HD + EPS);
        xn0 = x0 * rs * (1.0f + w[lane]);
        xn1 = x1 * rs * (1.0f + w[lane + 32]);
    }
    float o0 = xn0, o1 = xn1;
    if (do_rope) {
        float c0 = cosT[(size_t)s * HD + lane],      s0 = sinT[(size_t)s * HD + lane];
        float c1 = cosT[(size_t)s * HD + lane + 32], s1 = sinT[(size_t)s * HD + lane + 32];
        o0 = xn0 * c0 - xn1 * s0;
        o1 = xn1 * c1 + xn0 * s1;
    }
    if (transpose_out) {
        size_t ob = (size_t)(b * nh + hh) * HD;
        out[(ob + lane) * Ss + s]      = (_Float16)(o0 * scale);
        out[(ob + lane + 32) * Ss + s] = (_Float16)(o1 * scale);
    } else {
        size_t ob = ((size_t)(b * nh + hh) * Ss + s) * HD;
        out[ob + lane]      = (_Float16)(o0 * scale);
        out[ob + lane + 32] = (_Float16)(o1 * scale);
    }
}

// ---------------- silu(a)*b -> f16 ----------------
__global__ __launch_bounds__(256) void nordic_gate_kernel(const float* __restrict__ a,
                                                          const float* __restrict__ b,
                                                          _Float16* __restrict__ g) {
    size_t i = (size_t)blockIdx.x * 256 + threadIdx.x;
    if (i >= (size_t)Tt * Ff) return;
    float av = a[i];
    g[i] = (_Float16)((av / (1.0f + __expf(-av))) * b[i]);
}

// ---------------- WMMA GEMM: C[MxN] = A[MxK] * B^T, with B stored [N x K] ----------------
// one wave computes a 32x64 tile of C: 2 M-tiles x 4 N-tiles share fragments
__global__ __launch_bounds__(256) void nordic_gemm_kernel(const _Float16* __restrict__ A,
                                                          const _Float16* __restrict__ B,
                                                          float* __restrict__ C,
                                                          int M, int N, int K,
                                                          int totalWaves) {
    int wid = blockIdx.x * (blockDim.x >> 5) + (threadIdx.x >> 5);
    if (wid >= totalWaves) return;
    int lane = threadIdx.x & 31;
    int ntiles = N >> 6;
    int mt = wid / ntiles, nt = wid % ntiles;
    int m0 = mt << 5, n0 = nt << 6;
    int r = lane & 15, hf = lane >> 4;
    int abase = hf * 8;

    v8f acc00{}, acc01{}, acc02{}, acc03{};
    v8f acc10{}, acc11{}, acc12{}, acc13{};

    const _Float16* Ar0 = A + (size_t)(m0 + r) * K;
    const _Float16* Ar1 = Ar0 + (size_t)16 * K;
    const _Float16* Br0 = B + (size_t)(n0 + r) * K;       // + t*16*K per N-tile

    for (int k0 = 0; k0 < K; k0 += 32) {
        // WGP-scope prefetch (locality 3): pull into all cache levels
        __builtin_prefetch(Ar0 + k0 + 128, 0, 3);
        __builtin_prefetch(Ar1 + k0 + 128, 0, 3);
        Frag a0, a1;
        {
            const v4f* p0 = (const v4f*)(Ar0 + k0);
            a0.f4[0] = p0[hf];  a0.f4[1] = p0[2 + hf];
            const v4f* p1 = (const v4f*)(Ar1 + k0);
            a1.f4[0] = p1[hf];  a1.f4[1] = p1[2 + hf];
        }
#pragma unroll
        for (int t = 0; t < 4; ++t) {
            Frag b;
            const v4f* pb = (const v4f*)(Br0 + (size_t)t * 16 * K + k0);
            b.f4[0] = pb[hf];  b.f4[1] = pb[2 + hf];
            v8f c0 = (t == 0) ? acc00 : (t == 1) ? acc01 : (t == 2) ? acc02 : acc03;
            v8f c1 = (t == 0) ? acc10 : (t == 1) ? acc11 : (t == 2) ? acc12 : acc13;
            c0 = __builtin_amdgcn_wmma_f32_16x16x32_f16(false, a0.v, false, b.v, (short)0, c0, false, false);
            c1 = __builtin_amdgcn_wmma_f32_16x16x32_f16(false, a1.v, false, b.v, (short)0, c1, false, false);
            if (t == 0) { acc00 = c0; acc10 = c1; }
            else if (t == 1) { acc01 = c0; acc11 = c1; }
            else if (t == 2) { acc02 = c0; acc12 = c1; }
            else { acc03 = c0; acc13 = c1; }
        }
    }
#pragma unroll
    for (int t = 0; t < 4; ++t) {
        v8f c0 = (t == 0) ? acc00 : (t == 1) ? acc01 : (t == 2) ? acc02 : acc03;
        v8f c1 = (t == 0) ? acc10 : (t == 1) ? acc11 : (t == 2) ? acc12 : acc13;
        int n = n0 + t * 16 + r;
#pragma unroll
        for (int j = 0; j < 8; ++j) {
            C[(size_t)(m0 + j + abase) * N + n] = c0[j];
            C[(size_t)(m0 + 16 + j + abase) * N + n] = c1[j];
        }
    }
}

// ---------------- WMMA flash attention ----------------
// one wave per (b, h, 16-row q tile); q16:(b,H,s,hd)  k16:(b,G,s,hd)  vT:(b,G,hd,s)  ctx16:(b,s,H,hd)
__global__ __launch_bounds__(128) void nordic_attn_kernel(const _Float16* __restrict__ q16,
                                                          const _Float16* __restrict__ k16,
                                                          const _Float16* __restrict__ vT,
                                                          _Float16* __restrict__ ctx16,
                                                          int local) {
    __shared__ _Float16 lds[4 * 16 * 32];
    int w = threadIdx.x >> 5;
    int wid = blockIdx.x * 4 + w;
    int lane = threadIdx.x & 31;
    const int QT = Ss / 16;
    int qt = wid % QT;
    int h  = (wid / QT) % Hh;
    int b  = wid / (QT * Hh);
    int g  = h / GROUP;
    int i0 = qt * 16;
    int r = lane & 15, hf = lane >> 4;
    int abase = hf * 8;
    _Float16* myl = lds + w * (16 * 32);

    // q A-fragments for hd chunks [0,32) and [32,64)
    Frag aq0, aq1;
    {
        const _Float16* qrow = q16 + (size_t)((b * Hh + h) * Ss + i0 + r) * HD;
        const v4f* p0 = (const v4f*)qrow;
        const v4f* p1 = (const v4f*)(qrow + 32);
        aq0.f4[0] = p0[hf];  aq0.f4[1] = p0[2 + hf];
        aq1.f4[0] = p1[hf];  aq1.f4[1] = p1[2 + hf];
    }

    v8f o0{}, o1{}, o2{}, o3{};
    float mrow[8], lsum[8];
#pragma unroll
    for (int j = 0; j < 8; ++j) { mrow[j] = -3.0e38f; lsum[j] = 0.f; }

    const _Float16* kb_p = k16 + (size_t)((b * Gg + g) * Ss) * HD;
    const _Float16* vb_p = vT + (size_t)((b * Gg + g) * HD) * Ss;

    int kend = i0 + 16;
    int kst = 0;
    if (local) { kst = i0 - WINDOW; if (kst < 0) kst = 0; kst &= ~31; }

    for (int k0 = kst; k0 < kend; k0 += 32) {
        // ---- scores for 32 keys (two 16x16 tiles) ----
        v8f st0{}, st1{};
        {
            const v4f* pk0 = (const v4f*)(kb_p + (size_t)(k0 + r) * HD);
            Frag bk;
            bk.f4[0] = pk0[hf];      bk.f4[1] = pk0[2 + hf];
            st0 = __builtin_amdgcn_wmma_f32_16x16x32_f16(false, aq0.v, false, bk.v, (short)0, st0, false, false);
            bk.f4[0] = pk0[4 + hf];  bk.f4[1] = pk0[6 + hf];
            st0 = __builtin_amdgcn_wmma_f32_16x16x32_f16(false, aq1.v, false, bk.v, (short)0, st0, false, false);
            const v4f* pk1 = (const v4f*)(kb_p + (size_t)(k0 + 16 + r) * HD);
            bk.f4[0] = pk1[hf];      bk.f4[1] = pk1[2 + hf];
            st1 = __builtin_amdgcn_wmma_f32_16x16x32_f16(false, aq0.v, false, bk.v, (short)0, st1, false, false);
            bk.f4[0] = pk1[4 + hf];  bk.f4[1] = pk1[6 + hf];
            st1 = __builtin_amdgcn_wmma_f32_16x16x32_f16(false, aq1.v, false, bk.v, (short)0, st1, false, false);
        }
        // ---- mask (causal + optional sliding window) ----
        {
            int jc0 = k0 + r, jc1 = k0 + 16 + r;
#pragma unroll
            for (int j = 0; j < 8; ++j) {
                int irow = i0 + j + abase;
                if ((jc0 > irow) || (local && (irow - jc0 > WINDOW))) st0[j] = -1.0e30f;
                if ((jc1 > irow) || (local && (irow - jc1 > WINDOW))) st1[j] = -1.0e30f;
            }
        }
        // ---- online softmax ----
        float pn0[8], pn1[8];
#pragma unroll
        for (int j = 0; j < 8; ++j) {
            float rm = fmaxf(st0[j], st1[j]);
            rm = fmaxf(rm, __shfl_xor(rm, 1, 32));
            rm = fmaxf(rm, __shfl_xor(rm, 2, 32));
            rm = fmaxf(rm, __shfl_xor(rm, 4, 32));
            rm = fmaxf(rm, __shfl_xor(rm, 8, 32));
            float mnew  = fmaxf(mrow[j], rm);
            float alpha = __expf(mrow[j] - mnew);
            mrow[j] = mnew;
            lsum[j] *= alpha;
            o0[j] *= alpha; o1[j] *= alpha; o2[j] *= alpha; o3[j] *= alpha;
            float p0 = (st0[j] <= -1.0e29f) ? 0.f : __expf(st0[j] - mnew);
            float p1 = (st1[j] <= -1.0e29f) ? 0.f : __expf(st1[j] - mnew);
            float rs = p0 + p1;
            rs += __shfl_xor(rs, 1, 32); rs += __shfl_xor(rs, 2, 32);
            rs += __shfl_xor(rs, 4, 32); rs += __shfl_xor(rs, 8, 32);
            lsum[j] += rs;
            pn0[j] = p0; pn1[j] = p1;
        }
        // ---- P (C-layout) -> LDS -> A-layout fragment ----
#pragma unroll
        for (int j = 0; j < 8; ++j) {
            int row = j + abase;
            myl[row * 32 + r]      = (_Float16)pn0[j];
            myl[row * 32 + 16 + r] = (_Float16)pn1[j];
        }
        asm volatile("s_wait_dscnt 0" ::: "memory");
        Frag pa;
        {
            const v4f* pp = (const v4f*)(myl + r * 32);
            pa.f4[0] = pp[hf];  pa.f4[1] = pp[2 + hf];
        }
        // ---- O += P @ V (contraction over 32 keys, 4 hd tiles; V transposed -> contiguous) ----
#pragma unroll
        for (int t = 0; t < 4; ++t) {
            Frag bv;
            const v4f* pv = (const v4f*)(vb_p + (size_t)(t * 16 + r) * Ss + k0);
            bv.f4[0] = pv[hf];  bv.f4[1] = pv[2 + hf];
            v8f oo = (t == 0) ? o0 : (t == 1) ? o1 : (t == 2) ? o2 : o3;
            oo = __builtin_amdgcn_wmma_f32_16x16x32_f16(false, pa.v, false, bv.v, (short)0, oo, false, false);
            if (t == 0) o0 = oo; else if (t == 1) o1 = oo; else if (t == 2) o2 = oo; else o3 = oo;
        }
    }
    // ---- normalize + store ctx in (b, s, H, hd) ----
#pragma unroll
    for (int t = 0; t < 4; ++t) {
        v8f oo = (t == 0) ? o0 : (t == 1) ? o1 : (t == 2) ? o2 : o3;
#pragma unroll
        for (int j = 0; j < 8; ++j) {
            int m = j + abase;
            ctx16[((size_t)((b * Ss + i0 + m) * Hh + h)) * HD + t * 16 + r] =
                (_Float16)(oo[j] / lsum[j]);
        }
    }
}

// ---------------- host launcher ----------------
extern "C" void kernel_launch(void* const* d_in, const int* in_sizes, int n_in,
                              void* d_out, int out_size, void* d_ws, size_t ws_size,
                              hipStream_t stream) {
    (void)in_sizes; (void)n_in; (void)out_size; (void)ws_size;

    const int*   ids    = (const int*)d_in[0];
    const float* tokemb = (const float*)d_in[1];
    const float* Wq     = (const float*)d_in[2];
    const float* Wk     = (const float*)d_in[3];
    const float* Wv     = (const float*)d_in[4];
    const float* Wo     = (const float*)d_in[5];
    const float* W1     = (const float*)d_in[6];
    const float* W2     = (const float*)d_in[7];
    const float* W3     = (const float*)d_in[8];
    const float* n1     = (const float*)d_in[9];
    const float* n2     = (const float*)d_in[10];
    const float* n3     = (const float*)d_in[11];
    const float* n4     = (const float*)d_in[12];
    const float* qn     = (const float*)d_in[13];
    const float* kn     = (const float*)d_in[14];
    const float* nf     = (const float*)d_in[15];
    const float* cosT   = (const float*)d_in[16];
    const float* sinT   = (const float*)d_in[17];
    float* logits = (float*)d_out;

    // ---- workspace carve-up ----
    size_t off = 0;
    auto carve = [&](size_t bytes) -> char* {
        char* p = (char*)d_ws + off;
        off += (bytes + 255) & ~(size_t)255;
        return p;
    };
    _Float16* emb16 = (_Float16*)carve((size_t)Vv * Dd * 2);               // [V x D] (already B^T layout)
    _Float16* wq16  = (_Float16*)carve((size_t)Ll * Dd * (Hh * HD) * 2);   // [N x K] per layer
    _Float16* wk16  = (_Float16*)carve((size_t)Ll * Dd * (Gg * HD) * 2);
    _Float16* wv16  = (_Float16*)carve((size_t)Ll * Dd * (Gg * HD) * 2);
    _Float16* wo16  = (_Float16*)carve((size_t)Ll * (Hh * HD) * Dd * 2);
    _Float16* w116  = (_Float16*)carve((size_t)Ll * Dd * Ff * 2);
    _Float16* w216  = (_Float16*)carve((size_t)Ll * Dd * Ff * 2);
    _Float16* w316  = (_Float16*)carve((size_t)Ll * Ff * Dd * 2);
    float*    x     = (float*)carve((size_t)Tt * Dd * 4);
    _Float16* h16   = (_Float16*)carve((size_t)Tt * Dd * 2);
    float*    qf    = (float*)carve((size_t)Tt * (Hh * HD) * 4);
    float*    kf    = (float*)carve((size_t)Tt * (Gg * HD) * 4);
    float*    vf    = (float*)carve((size_t)Tt * (Gg * HD) * 4);
    _Float16* q16   = (_Float16*)carve((size_t)Tt * (Hh * HD) * 2);
    _Float16* k16   = (_Float16*)carve((size_t)Tt * (Gg * HD) * 2);
    _Float16* v16T  = (_Float16*)carve((size_t)Tt * (Gg * HD) * 2);
    _Float16* ctx16 = (_Float16*)carve((size_t)Tt * (Hh * HD) * 2);
    float*    attn  = (float*)carve((size_t)Tt * Dd * 4);
    float*    ff1   = (float*)carve((size_t)Tt * Ff * 4);
    float*    ff2   = (float*)carve((size_t)Tt * Ff * 4);
    _Float16* g16   = (_Float16*)carve((size_t)Tt * Ff * 2);
    float*    ffo   = (float*)carve((size_t)Tt * Dd * 4);

    auto convT = [&](const float* in, _Float16* out, int K, int N) {
        dim3 grid(N / 32, K / 32, Ll);
        nordic_convT_kernel<<<grid, 256, 0, stream>>>(in, out, K, N);
    };
    auto gemm = [&](const _Float16* A, const _Float16* Bm, float* C, int M, int N, int K) {
        int waves = (M / 32) * (N / 64);
        nordic_gemm_kernel<<<(waves + 7) / 8, 256, 0, stream>>>(A, Bm, C, M, N, K, waves);
    };

    // ---- weight conversion (+transpose to [N x K]) ----
    {
        size_t n = (size_t)Vv * Dd;
        nordic_conv_kernel<<<(unsigned)((n + 255) / 256), 256, 0, stream>>>(tokemb, emb16, n);
    }
    convT(Wq, wq16, Dd, Hh * HD);
    convT(Wk, wk16, Dd, Gg * HD);
    convT(Wv, wv16, Dd, Gg * HD);
    convT(Wo, wo16, Hh * HD, Dd);
    convT(W1, w116, Dd, Ff);
    convT(W2, w216, Dd, Ff);
    convT(W3, w316, Ff, Dd);

    // ---- embedding ----
    nordic_embed_kernel<<<(unsigned)(((size_t)Tt * Dd + 255) / 256), 256, 0, stream>>>(ids, tokemb, x);

    const int qkrows = Tt * Hh, kvrows = Tt * Gg;

    for (int l = 0; l < Ll; ++l) {
        const _Float16* lwq = wq16 + (size_t)l * Dd * (Hh * HD);
        const _Float16* lwk = wk16 + (size_t)l * Dd * (Gg * HD);
        const _Float16* lwv = wv16 + (size_t)l * Dd * (Gg * HD);
        const _Float16* lwo = wo16 + (size_t)l * (Hh * HD) * Dd;
        const _Float16* lw1 = w116 + (size_t)l * Dd * Ff;
        const _Float16* lw2 = w216 + (size_t)l * Dd * Ff;
        const _Float16* lw3 = w316 + (size_t)l * Ff * Dd;

        // pre-attn norm
        nordic_rmsnorm_kernel<<<Tt, 256, 0, stream>>>(x, n1 + (size_t)l * Dd, h16);
        // QKV projections
        gemm(h16, lwq, qf, Tt, Hh * HD, Dd);
        gemm(h16, lwk, kf, Tt, Gg * HD, Dd);
        gemm(h16, lwv, vf, Tt, Gg * HD, Dd);
        // qk norm + rope (+ q scale), v relayout (transposed for P@V)
        nordic_qkrope_kernel<<<(qkrows + 7) / 8, 256, 0, stream>>>(
            qf, qn + (size_t)l * HD, cosT, sinT, q16, Hh, 0.125f, 1, 0);
        nordic_qkrope_kernel<<<(kvrows + 7) / 8, 256, 0, stream>>>(
            kf, kn + (size_t)l * HD, cosT, sinT, k16, Gg, 1.0f, 1, 0);
        nordic_qkrope_kernel<<<(kvrows + 7) / 8, 256, 0, stream>>>(
            vf, nullptr, cosT, sinT, v16T, Gg, 1.0f, 0, 1);
        // attention
        {
            int waves = Bb * Hh * (Ss / 16);
            nordic_attn_kernel<<<waves / 4, 128, 0, stream>>>(q16, k16, v16T, ctx16,
                                                              (l < Ll - 4) ? 1 : 0);
        }
        // output projection + residual
        gemm(ctx16, lwo, attn, Tt, Dd, Hh * HD);
        nordic_addnorm_kernel<<<Tt, 256, 0, stream>>>(x, attn, n2 + (size_t)l * Dd);
        // FFN
        nordic_rmsnorm_kernel<<<Tt, 256, 0, stream>>>(x, n3 + (size_t)l * Dd, h16);
        gemm(h16, lw1, ff1, Tt, Ff, Dd);
        gemm(h16, lw2, ff2, Tt, Ff, Dd);
        nordic_gate_kernel<<<(unsigned)(((size_t)Tt * Ff + 255) / 256), 256, 0, stream>>>(ff1, ff2, g16);
        gemm(g16, lw3, ffo, Tt, Dd, Ff);
        nordic_addnorm_kernel<<<Tt, 256, 0, stream>>>(x, ffo, n4 + (size_t)l * Dd);
    }

    // ---- final norm + logits (emb16 [V x D] is already the B^T layout) ----
    nordic_rmsnorm_kernel<<<Tt, 256, 0, stream>>>(x, nf, h16);
    gemm(h16, emb16, logits, Tt, Vv, Dd);
}
